// CrossAttention_42880953483266
// MI455X (gfx1250) — compile-verified, matrix operand order checked by reading, not run on hardware
//
#include <hip/hip_runtime.h>

#define NB 4
#define SQL 1024
#define SKV 2048
#define DM 1024
#define NH 16
#define HD 64

typedef __attribute__((ext_vector_type(16))) __bf16 v16bf;
typedef __attribute__((ext_vector_type(8)))  __bf16 v8bf;
typedef __attribute__((ext_vector_type(8)))  float  v8f;
typedef __attribute__((ext_vector_type(4)))  unsigned int u32x4;
typedef __attribute__((ext_vector_type(8)))  unsigned int u32x8;

// Load a 16-bit A/B WMMA fragment per CDNA5 ISA layout:
// lane half (lane>>4) selects K sub-chunks [half*8, half*8+8) and [16+half*8, ...+8)
__device__ __forceinline__ v16bf frag16(const __bf16* p, int half) {
  v8bf lo = *(const v8bf*)(p + half * 8);
  v8bf hi = *(const v8bf*)(p + 16 + half * 8);
  return __builtin_shufflevector(lo, hi, 0,1,2,3,4,5,6,7,8,9,10,11,12,13,14,15);
}

__device__ __forceinline__ v8f wmma_bf16(v16bf a, v16bf b, v8f c) {
  return __builtin_amdgcn_wmma_f32_16x16x32_bf16(false, a, false, b, (short)0, c, false, false);
}

// ---------------------------------------------------------------------------
// Generic 64x64-tile GEMM, bf16 WMMA, f32 accumulate.
// MODE 0: A=f32 hidden_states, out -> Qh bf16 (B,H,SQ,HD)
// MODE 1: A=f32 encoder_hidden_states, out -> Kh (B,H,SKV,HD) / VhT (B,H,HD,SKV)
// MODE 2: A=bf16 attn head output (B,SQ,D), out -> f32 d_out attn_output
// ---------------------------------------------------------------------------
template<int MODE>
__global__ void __launch_bounds__(256)
gemm_wmma(const float* __restrict__ Af, const __bf16* __restrict__ Ab,
          const float* __restrict__ Bw, const float* __restrict__ bias,
          __bf16* __restrict__ outK, __bf16* __restrict__ outV,
          float* __restrict__ outF, int M, int N, int K)
{
  __shared__ __bf16 As[64][32];   // A tile, row-major (m, k)
  __shared__ __bf16 Bt[64][32];   // B tile transposed (n, k)

  const int tid  = threadIdx.x;
  const int wave = tid >> 5;
  const int lane = tid & 31;
  const int mBase = blockIdx.y * 64;
  const int nBase = blockIdx.x * 64;
  const int mTile = wave & 3;          // 4 M-tiles
  const int nt0   = (wave >> 2) * 2;   // each wave owns 2 N-tiles

  v8f acc0 = {}; v8f acc1 = {};

  for (int kk = 0; kk < K; kk += 32) {
    // stage A (64x32): thread -> 8 contiguous elements
    {
      const int r = tid >> 2, c = (tid & 3) * 8;
      if (MODE == 2) {
        const __bf16* src = Ab + (size_t)(mBase + r) * K + kk + c;
        #pragma unroll
        for (int j = 0; j < 8; ++j) As[r][c + j] = src[j];
      } else {
        const float* src = Af + (size_t)(mBase + r) * K + kk + c;
        #pragma unroll
        for (int j = 0; j < 8; ++j) As[r][c + j] = (__bf16)src[j];
      }
    }
    // stage B transposed: thread loads 8 consecutive n for one k
    {
      const int kq = tid & 31, n8 = (tid >> 5) * 8;
      const float* src = Bw + (size_t)(kk + kq) * N + nBase + n8;
      #pragma unroll
      for (int j = 0; j < 8; ++j) Bt[n8 + j][kq] = (__bf16)src[j];
    }
    __syncthreads();

    const int half = lane >> 4;
    v16bf aF  = frag16(&As[mTile * 16 + (lane & 15)][0], half);
    v16bf bF0 = frag16(&Bt[nt0 * 16       + (lane & 15)][0], half);
    v16bf bF1 = frag16(&Bt[(nt0 + 1) * 16 + (lane & 15)][0], half);
    acc0 = wmma_bf16(aF, bF0, acc0);
    acc1 = wmma_bf16(aF, bF1, acc1);
    __syncthreads();
  }

  // writeback: acc element i -> row = i + 8*(lane>>4), col = lane&15
  const int c0 = nt0 * 16 + (lane & 15);
  const int c1 = c0 + 16;
  const int gn0 = nBase + c0, gn1 = nBase + c1;
  const float b0 = bias[gn0], b1 = bias[gn1];
  #pragma unroll
  for (int i = 0; i < 8; ++i) {
    const int gm = mBase + mTile * 16 + (lane >> 4) * 8 + i;
    const float v0 = acc0[i] + b0;
    const float v1 = acc1[i] + b1;
    if (MODE == 0) {
      const int b = gm >> 10, sq = gm & 1023;
      outK[(((size_t)(b * NH + (gn0 >> 6)) * SQL + sq) << 6) + (gn0 & 63)] = (__bf16)v0;
      outK[(((size_t)(b * NH + (gn1 >> 6)) * SQL + sq) << 6) + (gn1 & 63)] = (__bf16)v1;
    } else if (MODE == 1) {
      const int b = gm >> 11, skv = gm & 2047;
      if (gn0 < DM)
        outK[(((size_t)(b * NH + (gn0 >> 6)) * SKV + skv) << 6) + (gn0 & 63)] = (__bf16)v0;
      else   // V stored transposed per head: (B,H,HD,SKV)
        outV[((size_t)(b * NH + ((gn0 - DM) >> 6)) * HD + ((gn0 - DM) & 63)) * SKV + skv] = (__bf16)v0;
      if (gn1 < DM)
        outK[(((size_t)(b * NH + (gn1 >> 6)) * SKV + skv) << 6) + (gn1 & 63)] = (__bf16)v1;
      else
        outV[((size_t)(b * NH + ((gn1 - DM) >> 6)) * HD + ((gn1 - DM) & 63)) * SKV + skv] = (__bf16)v1;
    } else {
      outF[(size_t)gm * DM + gn0] = v0;
      outF[(size_t)gm * DM + gn1] = v1;
    }
  }
}

// ---------------------------------------------------------------------------
// Attention: one block (8 waves) per (b, h, 16 q-rows).
// Full 16x2048 f32 score strip + 16x2048 bf16 prob strip in LDS (320KB/WGP).
// Q tile (16x64 bf16) brought in via the Tensor Data Mover.
// ---------------------------------------------------------------------------
__global__ void __launch_bounds__(256)
attn_wmma(const __bf16* __restrict__ Qh, const __bf16* __restrict__ Kh,
          const __bf16* __restrict__ VhT, const float* __restrict__ mask,
          float* __restrict__ attnW, __bf16* __restrict__ Oh)
{
  extern __shared__ char smem[];
  float*  S    = (float*)smem;                            // 16 x 2048 f32   (128K)
  __bf16* Pb   = (__bf16*)(smem + 131072);                // 16 x 2048 bf16  ( 64K)
  float*  Oacc = (float*)(smem + 131072 + 65536);         // 16 x 64 f32     (  4K)
  float*  red  = Oacc + 16 * 64;                          // 272 f32
  __bf16* Qs   = (__bf16*)(smem + 131072 + 65536 + 4096 + 1088); // 16 x 64 bf16 (2K)

  const int tid  = threadIdx.x;
  const int wave = tid >> 5;
  const int lane = tid & 31;
  const int blk  = blockIdx.x;
  const int qb   = blk & ((SQL / 16) - 1);
  const int bh   = blk >> 6;               // b*NH + h
  const int b    = bh >> 4;
  const int h    = bh & 15;

  for (int i = tid; i < 16 * 64; i += 256) Oacc[i] = 0.f;

  // ---- TDM: pull the 16x64 bf16 Q tile into LDS with tensor_load_to_lds ----
  if (wave == 0) {
    const unsigned long long gaddr =
        (unsigned long long)(Qh + ((size_t)bh * SQL + qb * 16) * HD);
    const unsigned lds_addr = (unsigned)(unsigned long long)Qs;
    u32x4 g0;
    g0[0] = 1u;                                        // count=1 (valid D#)
    g0[1] = lds_addr;                                  // lds_addr
    g0[2] = (unsigned)gaddr;                           // global_addr[31:0]
    g0[3] = ((unsigned)(gaddr >> 32) & 0x01FFFFFFu) | 0x80000000u; // addr[56:32] | type=2
    u32x8 g1;
    g1[0] = 1u << 16;        // data_size = 1 (2 bytes)
    g1[1] = 64u << 16;       // tensor_dim0 = 64 (bits 79:48, low half)
    g1[2] = 16u << 16;       // tensor_dim1 = 16 (bits 111:80, low half)
    g1[3] = 64u << 16;       // tile_dim0 = 64 (bits 127:112)
    g1[4] = 16u;             // tile_dim1 = 16, tile_dim2 = 0
    g1[5] = 64u;             // tensor_dim0_stride = 64 (bits 207:160)
    g1[6] = 0u;              // stride hi / tensor_dim1_stride
    g1[7] = 0u;
    u32x4 g2 = {0u, 0u, 0u, 0u};
    u32x4 g3 = {0u, 0u, 0u, 0u};
    asm volatile("tensor_load_to_lds %0, %1, %2, %3"
                 :: "s"(g0), "s"(g1), "s"(g2), "s"(g3) : "memory");
    __builtin_amdgcn_s_wait_tensorcnt(0);
  }
  __syncthreads();

  // ---- Phase 1: S = (Q K^T) * rsqrt(HD) + mask ----
  const int half = lane >> 4;
  v16bf qF0 = frag16(&Qs[(lane & 15) * HD], half);
  v16bf qF1 = frag16(&Qs[(lane & 15) * HD + 32], half);

  #pragma unroll 4
  for (int t = 0; t < 16; ++t) {            // 16 N-tiles -> 256 KV cols per wave
    const int nB = wave * 256 + t * 16;
    const __bf16* krow = Kh + ((size_t)bh * SKV + nB + (lane & 15)) * HD;
    v16bf kF0 = frag16(krow, half);
    v16bf kF1 = frag16(krow + 32, half);
    v8f acc = {};
    acc = wmma_bf16(qF0, kF0, acc);
    acc = wmma_bf16(qF1, kF1, acc);
    const int col = nB + (lane & 15);
    const float mv = mask[b * SKV + col];
    #pragma unroll
    for (int i = 0; i < 8; ++i) {
      const int row = (lane >> 4) * 8 + i;
      S[row * SKV + col] = acc[i] * 0.125f + mv;
    }
  }
  __syncthreads();

  // ---- Phase 2: exact softmax; stream probs to d_out, bf16 copy to LDS ----
  {
    const int row = tid >> 4, sub = tid & 15;
    float mx = -3.4e38f;
    for (int c = sub; c < SKV; c += 16) mx = fmaxf(mx, S[row * SKV + c]);
    red[row * 16 + sub] = mx;
    __syncthreads();
    if (sub == 0) {
      float m2 = red[row * 16];
      #pragma unroll
      for (int j = 1; j < 16; ++j) m2 = fmaxf(m2, red[row * 16 + j]);
      red[256 + row] = m2;
    }
    __syncthreads();
    const float rmax = red[256 + row];
    float sum = 0.f;
    for (int c = sub; c < SKV; c += 16) {
      const float e = __expf(S[row * SKV + c] - rmax);
      S[row * SKV + c] = e;
      sum += e;
    }
    __syncthreads();
    red[row * 16 + sub] = sum;
    __syncthreads();
    if (sub == 0) {
      float s2 = 0.f;
      #pragma unroll
      for (int j = 0; j < 16; ++j) s2 += red[row * 16 + j];
      red[256 + row] = s2;
    }
    __syncthreads();
    const float inv = 1.f / red[256 + row];
    const size_t wOff = ((size_t)bh * SQL + qb * 16 + row) * SKV;
    for (int c = sub; c < SKV; c += 16) {
      const float p = S[row * SKV + c] * inv;
      Pb[row * SKV + c] = (__bf16)p;
      attnW[wOff + c] = p;
    }
  }
  __syncthreads();

  // ---- Phase 3: O = P @ V. B-frags stream directly from transposed V. ----
  const int nTile = wave & 3;     // hd 16-block
  const int kHalf = wave >> 2;    // split K=2048 across wave pairs
  const __bf16* vrow =
      VhT + ((size_t)bh * HD + nTile * 16 + (lane & 15)) * SKV + kHalf * 1024;
  const __bf16* prow = &Pb[(lane & 15) * SKV + kHalf * 1024];
  v8f oacc = {};
  #pragma unroll 4
  for (int k0 = 0; k0 < 1024; k0 += 32) {
    v16bf aF = frag16(prow + k0, half);
    v16bf bF = frag16(vrow + k0, half);
    oacc = wmma_bf16(aF, bF, oacc);
  }
  #pragma unroll
  for (int i = 0; i < 8; ++i) {
    const int row = (lane >> 4) * 8 + i;
    const int cc  = nTile * 16 + (lane & 15);
    atomicAdd(&Oacc[row * 64 + cc], oacc[i]);   // ds_add_f32 cross-wave reduce
  }
  __syncthreads();
  // write head output in (B,SQ,D) layout for the final projection GEMM
  for (int i = tid; i < 16 * 64; i += 256) {
    const int row = i >> 6, cc = i & 63;
    const int sq = qb * 16 + row;
    Oh[((size_t)(b * SQL + sq)) * DM + h * HD + cc] = (__bf16)Oacc[i];
  }
}

// ---------------------------------------------------------------------------
extern "C" void kernel_launch(void* const* d_in, const int* in_sizes, int n_in,
                              void* d_out, int out_size, void* d_ws, size_t ws_size,
                              hipStream_t stream) {
  const float* hs   = (const float*)d_in[0];
  const float* ehs  = (const float*)d_in[1];
  const float* mask = (const float*)d_in[2];
  const float* qw   = (const float*)d_in[3];
  const float* qb   = (const float*)d_in[4];
  const float* cw   = (const float*)d_in[5];
  const float* cb   = (const float*)d_in[6];
  const float* pw   = (const float*)d_in[7];
  const float* pb   = (const float*)d_in[8];

  float* attn_out = (float*)d_out;                          // B*SQ*D f32
  float* attn_w   = attn_out + (size_t)NB * SQL * DM;       // B*H*SQ*SKV f32

  char* ws = (char*)d_ws;
  __bf16* Qh  = (__bf16*)ws;                                //  8 MB (B,H,SQ,HD)
  __bf16* Kh  = (__bf16*)(ws + 8388608);                    // 16 MB (B,H,SKV,HD)
  __bf16* VhT = (__bf16*)(ws + 25165824);                   // 16 MB (B,H,HD,SKV)
  __bf16* Oh  = (__bf16*)(ws + 41943040);                   //  8 MB (B,SQ,D)

  dim3 blk(256);

  // Q projection: M=4096, N=1024, K=1024
  gemm_wmma<0><<<dim3(DM / 64, (NB * SQL) / 64), blk, 0, stream>>>(
      hs, nullptr, qw, qb, Qh, nullptr, nullptr, NB * SQL, DM, DM);

  // KV projection: M=8192, N=2048, K=1024
  gemm_wmma<1><<<dim3((2 * DM) / 64, (NB * SKV) / 64), blk, 0, stream>>>(
      ehs, nullptr, cw, cb, Kh, VhT, nullptr, NB * SKV, 2 * DM, DM);

  // Attention: one block per (b, h, 16 q-rows)
  const size_t shmem = 131072     // S (f32 scores)
                     + 65536      // Pb (bf16 probs)
                     + 4096       // Oacc
                     + 1088       // reductions
                     + 2048;      // Qs (TDM-loaded Q tile)
  attn_wmma<<<dim3(NB * NH * (SQL / 16)), blk, shmem, stream>>>(
      Qh, Kh, VhT, mask, attn_w, Oh);

  // Output projection: M=4096, N=1024, K=1024
  gemm_wmma<2><<<dim3(DM / 64, (NB * SQL) / 64), blk, 0, stream>>>(
      nullptr, Oh, pw, pb, nullptr, nullptr, attn_out, NB * SQL, DM, DM);
}
